// EEGMI_RWKV_ResNet_Model_48996986913353
// MI455X (gfx1250) — matrix operand order repaired and checked by reading, not compiled, not run on hardware
//
#include <hip/hip_runtime.h>
#include <hip/hip_bf16.h>

// ---------------- model dims ----------------
#define Bsz   32
#define Tn    2048
#define Cn    64
#define NBc   5
#define C5n   320
#define Hn    128
#define Lw    2      // RWKV layers
#define NBLKn 2
#define NCLSn 4
#define EPSf  1e-5f
#define RSn   (Tn + 16)   // padded time stride for bf16 feature maps (8-elem halo each side)

// ---------------- WMMA types ----------------
typedef __attribute__((ext_vector_type(16))) __bf16 v16bf;
typedef __attribute__((ext_vector_type(8)))  float  v8f;

union Frag {                 // bf16 fragment with 32-bit access for contiguous pairs
  v16bf v;
  unsigned int u32[8];
  unsigned short u16[16];
};

__device__ __forceinline__ unsigned short f2bf(float f) {
  unsigned int x = __float_as_uint(f);
  unsigned int r = x + 0x7fffu + ((x >> 16) & 1u);   // round-to-nearest-even
  return (unsigned short)(r >> 16);
}
__device__ __forceinline__ float sigm(float x) { return 1.f / (1.f + __expf(-x)); }

template <int ACT>
__device__ __forceinline__ float act_apply(float v) {
  if (ACT == 1) return fmaxf(v, 0.f);
  if (ACT == 2) return sigm(v);
  return v;
}

// =====================================================================
// Weight preparation
// =====================================================================
__global__ void prep_resw(const float* __restrict__ rw, unsigned short* __restrict__ wt) {
  int idx = blockIdx.x * 256 + threadIdx.x;
  const int total = NBLKn * 2 * C5n * C5n * 3;
  if (idx >= total) return;
  int k  = idx % 3;
  int ci = (idx / 3) % C5n;
  int co = (idx / (3 * C5n)) % C5n;
  int cv = idx / (3 * C5n * C5n);
  wt[((size_t)cv * 3 + k) * C5n * C5n + (size_t)co * C5n + ci] = f2bf(rw[idx]);
}

__global__ void prep_bn(const float* __restrict__ g, const float* __restrict__ b,
                        const float* __restrict__ m, const float* __restrict__ v,
                        float* __restrict__ scale, float* __restrict__ bias) {
  int idx = blockIdx.x * 256 + threadIdx.x;
  if (idx >= NBLKn * 2 * C5n) return;
  float inv = g[idx] * rsqrtf(v[idx] + EPSf);
  scale[idx] = inv;
  bias[idx]  = b[idx] - m[idx] * inv;
}

__global__ void cvt_bf16(const float* __restrict__ s, unsigned short* __restrict__ d, int n) {
  int i = blockIdx.x * 256 + threadIdx.x;
  if (i < n) d[i] = f2bf(s[i]);
}

// zero the 8-element halos of both padded bf16 feature buffers
__global__ void pad_zero(unsigned short* __restrict__ a, unsigned short* __restrict__ b) {
  int idx = blockIdx.x * 256 + threadIdx.x;          // B*C5*16 threads
  if (idx >= Bsz * C5n * 16) return;
  int j   = idx & 15;
  int row = idx >> 4;
  size_t pos = (size_t)row * RSn + (j < 8 ? j : (RSn - 16 + j));
  a[pos] = 0;
  b[pos] = 0;
}

// =====================================================================
// Stage 1: depthwise band conv, pooling, attention, scaling
// =====================================================================
__global__ void dwconv(const float* __restrict__ x, const float* __restrict__ bw,
                       const float* __restrict__ bb, float* __restrict__ bands) {
  int t  = blockIdx.x * blockDim.x + threadIdx.x;
  int co = blockIdx.y;
  int b  = blockIdx.z;
  int c  = co / NBc;
  float acc = bb[co];
#pragma unroll
  for (int k = 0; k < 5; ++k) {
    int tt = t + k - 2;
    if (tt >= 0 && tt < Tn)
      acc += bw[co * 5 + k] * x[((size_t)b * Tn + tt) * Cn + c];
  }
  bands[((size_t)b * C5n + co) * Tn + t] = acc;
}

__global__ void pool_bc(const float* __restrict__ bands, float* __restrict__ pooled) {
  int c = blockIdx.x, b = blockIdx.y;
  const float* p = bands + ((size_t)b * C5n + (size_t)c * NBc) * Tn;
  float s = 0.f;
  for (int i = threadIdx.x; i < NBc * Tn; i += 256) s += p[i];
  __shared__ float red[256];
  red[threadIdx.x] = s; __syncthreads();
  for (int st = 128; st > 0; st >>= 1) {
    if (threadIdx.x < st) red[threadIdx.x] += red[threadIdx.x + st];
    __syncthreads();
  }
  if (threadIdx.x == 0) pooled[b * Cn + c] = red[0] * (1.f / (NBc * Tn));
}

__global__ void attn_softmax(const float* __restrict__ pooled, const float* __restrict__ aw,
                             const float* __restrict__ ab, float* __restrict__ attn) {
  int b = blockIdx.x, c = threadIdx.x;   // 64 threads
  float z = ab[c];
  for (int i = 0; i < Cn; ++i) z += pooled[b * Cn + i] * aw[c * Cn + i];
  __shared__ float red[64];
  red[c] = z; __syncthreads();
  for (int st = 32; st > 0; st >>= 1) { if (c < st) red[c] = fmaxf(red[c], red[c + st]); __syncthreads(); }
  float mx = red[0]; __syncthreads();
  float e = __expf(z - mx);
  red[c] = e; __syncthreads();
  for (int st = 32; st > 0; st >>= 1) { if (c < st) red[c] += red[c + st]; __syncthreads(); }
  attn[b * Cn + c] = e / red[0];
}

__global__ void scale_bands(float* __restrict__ bands, unsigned short* __restrict__ featBf,
                            const float* __restrict__ attn) {
  size_t idx = (size_t)blockIdx.x * 256 + threadIdx.x;   // B*C5*T total
  size_t t   = idx % Tn;
  size_t co  = (idx / Tn) % C5n;
  size_t b   = idx / ((size_t)Tn * C5n);
  float v = bands[idx] * attn[b * Cn + co / NBc];
  bands[idx] = v;                                         // feat (f32, unpadded)
  featBf[(b * C5n + co) * RSn + 8 + t] = f2bf(v);         // feat (bf16, padded)
}

// =====================================================================
// Residual conv block: 3 shifted WMMA GEMMs, 16x64 tile per wave.
// X is time-padded (stride RSn, +8 halo) so shifted loads need no guards.
//   Out[b,co,t] = relu( scale[co]*sum_{k,ci} Wt[k][co][ci]*X[b,ci,t+k-1] + bias[co] (+resid) )
// HASRES: also add f32 residual and write f32 output (conv2); else bf16 only (conv1).
// =====================================================================
template <bool HASRES>
__global__ __launch_bounds__(32)
void res_conv_gemm(const unsigned short* __restrict__ Wt,   // [3][C5][C5] bf16
                   const unsigned short* __restrict__ Xbf,  // [B][C5][RSn] bf16 (padded)
                   const float* __restrict__ scale,
                   const float* __restrict__ bias,
                   const float* __restrict__ resid,         // [B][C5][T] f32 (HASRES)
                   float* __restrict__ outF,                // [B][C5][T] f32 (HASRES)
                   unsigned short* __restrict__ outBf) {    // [B][C5][RSn] bf16 (padded)
  const int l   = threadIdx.x;
  const int t0  = blockIdx.x * 64;                          // 4 N-tiles of 16
  const int co0 = blockIdx.y * 16;
  const int b   = blockIdx.z;
  const int lan = l & 15, hs = l >> 4;
  v8f acc[4] = {};
  const unsigned short* XbP = Xbf + (size_t)b * C5n * RSn + 8;
  for (int k = 0; k < 3; ++k) {
    const unsigned short* Wrow = Wt + ((size_t)k * C5n + (co0 + lan)) * C5n;
    const int tbase = t0 + lan + (k - 1);                   // halo makes -1/+1 safe
    for (int kc = 0; kc < C5n / 32; ++kc) {
      const int ci0 = kc * 32;
      Frag a;
#pragma unroll
      for (int vv = 0; vv < 8; ++vv) {
        int kk = ci0 + ((vv >> 2) << 4) + (hs << 3) + ((vv & 3) << 1);
        a.u32[vv] = *(const unsigned int*)(Wrow + kk);      // contiguous bf16 pair
      }
      Frag bm[4];
#pragma unroll
      for (int tt = 0; tt < 4; ++tt) {
        const int tcol = tbase + tt * 16;
#pragma unroll
        for (int j = 0; j < 8; ++j) {
          int kb = ci0 + ((j >> 2) << 4) + (hs << 3) + ((j & 3) << 1);
          bm[tt].u16[2 * j]     = XbP[(size_t)kb * RSn + tcol];
          bm[tt].u16[2 * j + 1] = XbP[(size_t)(kb + 1) * RSn + tcol];
        }
      }
#pragma unroll
      for (int tt = 0; tt < 4; ++tt)
        acc[tt] = __builtin_amdgcn_wmma_f32_16x16x32_bf16(false, a.v, false, bm[tt].v,
                                                          (short)0, acc[tt], false, false);
    }
  }
#pragma unroll
  for (int tt = 0; tt < 4; ++tt) {
    const int t = t0 + tt * 16 + lan;
#pragma unroll
    for (int r = 0; r < 8; ++r) {
      int co = co0 + (hs << 3) + r;
      float v = acc[tt][r] * scale[co] + bias[co];
      if (HASRES) {
        size_t idxF = ((size_t)b * C5n + co) * Tn + t;
        v += resid[idxF];
        v = fmaxf(v, 0.f);
        outF[idxF] = v;
      } else {
        v = fmaxf(v, 0.f);
      }
      outBf[((size_t)b * C5n + co) * RSn + 8 + t] = f2bf(v);
    }
  }
}

// =====================================================================
// Generic dense GEMM, NT 16-wide N-tiles per wave sharing one A fragment:
//   Out[b,n,o] = act( sum_k W[o,k]*X[b,n,k] + bias[o] )
//   B-fragment element [k][n] = X[b*bsX + (n0+n)*nstride + k*kstride]
// KCONTIG: kstride==1 (32-bit fragment assembly). BOUND: guard rows vs Mvalid.
// =====================================================================
template <int NT, int ACT, bool BOUND, bool KCONTIG, bool HASBF, bool HASBIAS>
__global__ __launch_bounds__(32)
void gemm_bf16(const unsigned short* __restrict__ W,   // [Mvalid][K] row-major bf16
               const unsigned short* __restrict__ X,
               const float* __restrict__ bias,
               float* __restrict__ outF,
               unsigned short* __restrict__ outBf,
               int K, int Mvalid, int ldo,
               long long bsX, long long bsO,
               int nstride, int kstride) {
  const int l  = threadIdx.x;
  const int n0 = blockIdx.x * 16 * NT;
  const int m0 = blockIdx.y * 16;
  const int b  = blockIdx.z;
  const int lan = l & 15, hs = l >> 4;
  v8f acc[NT] = {};
  const int mrow = m0 + lan;
  const bool mok = !BOUND || (mrow < Mvalid);
  const unsigned short* Wrow = W + (size_t)mrow * K;
  const unsigned short* Xb   = X + (size_t)b * bsX;
  for (int k0 = 0; k0 < K; k0 += 32) {
    Frag a;
#pragma unroll
    for (int vv = 0; vv < 8; ++vv) {
      int kk = k0 + ((vv >> 2) << 4) + (hs << 3) + ((vv & 3) << 1);
      if (BOUND) {
        a.u16[2 * vv]     = mok ? Wrow[kk] : (unsigned short)0;
        a.u16[2 * vv + 1] = mok ? Wrow[kk + 1] : (unsigned short)0;
      } else {
        a.u32[vv] = *(const unsigned int*)(Wrow + kk);
      }
    }
    Frag bm[NT];
#pragma unroll
    for (int tt = 0; tt < NT; ++tt) {
      const unsigned short* Xn = Xb + (size_t)(n0 + tt * 16 + lan) * nstride;
#pragma unroll
      for (int j = 0; j < 8; ++j) {
        int kb = k0 + ((j >> 2) << 4) + (hs << 3) + ((j & 3) << 1);
        if (KCONTIG) {
          bm[tt].u32[j] = *(const unsigned int*)(Xn + kb);
        } else {
          bm[tt].u16[2 * j]     = Xn[(size_t)kb * kstride];
          bm[tt].u16[2 * j + 1] = Xn[(size_t)(kb + 1) * kstride];
        }
      }
    }
#pragma unroll
    for (int tt = 0; tt < NT; ++tt)
      acc[tt] = __builtin_amdgcn_wmma_f32_16x16x32_bf16(false, a.v, false, bm[tt].v,
                                                        (short)0, acc[tt], false, false);
  }
#pragma unroll
  for (int tt = 0; tt < NT; ++tt) {
    const int n = n0 + tt * 16 + lan;
    if (!BOUND) {
      // 8 consecutive output channels per lane -> vector stores
      float o8[8];
#pragma unroll
      for (int r = 0; r < 8; ++r) {
        int o = m0 + (hs << 3) + r;
        float v = acc[tt][r];
        if (HASBIAS) v += bias[o];
        o8[r] = act_apply<ACT>(v);
      }
      size_t base = (size_t)b * bsO + (size_t)n * ldo + m0 + (hs << 3);
      float4* dst = (float4*)(outF + base);
      dst[0] = make_float4(o8[0], o8[1], o8[2], o8[3]);
      dst[1] = make_float4(o8[4], o8[5], o8[6], o8[7]);
      if (HASBF) {
        uint4 pk;
        pk.x = (unsigned int)f2bf(o8[0]) | ((unsigned int)f2bf(o8[1]) << 16);
        pk.y = (unsigned int)f2bf(o8[2]) | ((unsigned int)f2bf(o8[3]) << 16);
        pk.z = (unsigned int)f2bf(o8[4]) | ((unsigned int)f2bf(o8[5]) << 16);
        pk.w = (unsigned int)f2bf(o8[6]) | ((unsigned int)f2bf(o8[7]) << 16);
        *(uint4*)(outBf + base) = pk;
      }
    } else {
#pragma unroll
      for (int r = 0; r < 8; ++r) {
        int o = m0 + (hs << 3) + r;
        if (o < Mvalid) {
          float v = acc[tt][r];
          if (HASBIAS) v += bias[o];
          v = act_apply<ACT>(v);
          outF[(size_t)b * bsO + (size_t)n * ldo + o] = v;
        }
      }
    }
  }
}

// =====================================================================
// RWKV support kernels
// =====================================================================
__global__ void mean_T(const float* __restrict__ h, float* __restrict__ outF,
                       unsigned short* __restrict__ outBf) {
  int b = blockIdx.x, i = threadIdx.x;          // 128 threads
  const float* p = h + (size_t)b * Tn * Hn + i;
  float s = 0.f;
  for (int t = 0; t < Tn; ++t) s += p[(size_t)t * Hn];
  s *= (1.f / Tn);
  outF[b * Hn + i] = s;
  if (outBf) outBf[b * Hn + i] = f2bf(s);
}

__global__ void mix_kvr(const float* __restrict__ h, const float* __restrict__ xm,
                        const float* __restrict__ tmk, const float* __restrict__ tmv,
                        const float* __restrict__ tmr,
                        unsigned short* __restrict__ xk, unsigned short* __restrict__ xv,
                        unsigned short* __restrict__ xr) {
  size_t idx = (size_t)blockIdx.x * 256 + threadIdx.x;  // B*T*H
  int i   = (int)(idx % Hn);
  size_t b = idx / ((size_t)Tn * Hn);
  float x = h[idx], m = xm[b * Hn + i];
  float k = tmk[i], v = tmv[i], r = tmr[i];
  xk[idx] = f2bf(x * k + (1.f - k) * m);
  xv[idx] = f2bf(x * v + (1.f - v) * m);
  xr[idx] = f2bf(x * r + (1.f - r) * m);
}

__global__ void rwkv_scan(const float* __restrict__ kF, const float* __restrict__ vF,
                          const float* __restrict__ rF, unsigned short* __restrict__ out) {
  int tid = blockIdx.x * blockDim.x + threadIdx.x;      // B*H lanes
  int b = tid / Hn, i = tid % Hn;
  size_t base = (size_t)b * Tn * Hn + i;
  float alpha = 0.f;
  for (int t = 0; t < Tn; ++t) {
    size_t idx = base + (size_t)t * Hn;
    float s = sigm(kF[idx]) * vF[idx];
    float wkv = s + alpha;
    alpha = s + 0.9f * alpha;
    out[idx] = f2bf(rF[idx] * wkv);
  }
}

__device__ __forceinline__ float block_sum_128(float v, float* red, int i) {
  red[i] = v; __syncthreads();
#pragma unroll
  for (int st = 64; st > 0; st >>= 1) {
    if (i < st) red[i] += red[i + st];
    __syncthreads();
  }
  float r = red[0];
  __syncthreads();
  return r;
}

// out_new = y + relu(LN2(y)),  y = LN1(h + attnout)   (written back into h)
__global__ __launch_bounds__(128)
void ln_ff(float* __restrict__ h, const float* __restrict__ ao,
           const float* __restrict__ g1, const float* __restrict__ b1,
           const float* __restrict__ g2, const float* __restrict__ b2) {
  size_t row = blockIdx.x;
  int i = threadIdx.x;
  size_t idx = row * Hn + i;
  __shared__ float red[128];
  float y0 = h[idx] + ao[idx];
  float mu  = block_sum_128(y0, red, i) * (1.f / Hn);
  float var = block_sum_128(y0 * y0, red, i) * (1.f / Hn) - mu * mu;
  float y = (y0 - mu) * rsqrtf(var + EPSf) * g1[i] + b1[i];
  float mu2  = block_sum_128(y, red, i) * (1.f / Hn);
  float var2 = block_sum_128(y * y, red, i) * (1.f / Hn) - mu2 * mu2;
  float z = (y - mu2) * rsqrtf(var2 + EPSf) * g2[i] + b2[i];
  h[idx] = y + fmaxf(z, 0.f);
}

// =====================================================================
// Host-side orchestration
// =====================================================================
extern "C" void kernel_launch(void* const* d_in, const int* in_sizes, int n_in,
                              void* d_out, int out_size, void* d_ws, size_t ws_size,
                              hipStream_t stream) {
  const float* x      = (const float*)d_in[0];
  const float* band_w = (const float*)d_in[1];
  const float* band_b = (const float*)d_in[2];
  const float* attn_w = (const float*)d_in[3];
  const float* attn_b = (const float*)d_in[4];
  const float* res_w  = (const float*)d_in[5];
  const float* bn_g   = (const float*)d_in[6];
  const float* bn_b   = (const float*)d_in[7];
  const float* bn_m   = (const float*)d_in[8];
  const float* bn_v   = (const float*)d_in[9];
  const float* proj_w = (const float*)d_in[10];
  const float* proj_b = (const float*)d_in[11];
  const float* tmk    = (const float*)d_in[12];
  const float* tmv    = (const float*)d_in[13];
  const float* tmr    = (const float*)d_in[14];
  const float* wk     = (const float*)d_in[15];
  const float* wv     = (const float*)d_in[16];
  const float* wr     = (const float*)d_in[17];
  const float* wo     = (const float*)d_in[18];
  const float* ln1g   = (const float*)d_in[19];
  const float* ln1b   = (const float*)d_in[20];
  const float* ln2g   = (const float*)d_in[21];
  const float* ln2b   = (const float*)d_in[22];
  const float* cls_w1 = (const float*)d_in[23];
  const float* cls_b1 = (const float*)d_in[24];
  const float* cls_w2 = (const float*)d_in[25];
  const float* cls_b2 = (const float*)d_in[26];

  char* ws = (char*)d_ws;

  // ---- region R0 (~168.4 MB), reused across the two phases ----
  const size_t SZ_BANDS = (size_t)Bsz * C5n * Tn * 4;     //  83,886,080
  const size_t SZ_FPAD  = (size_t)Bsz * C5n * RSn * 2;    //  42,270,720
  const size_t SZ_R0    = SZ_BANDS + 2 * SZ_FPAD;         // 168,427,520 (256-aligned)
  // phase 1: conv stage
  float*          bandsF = (float*)(ws);
  unsigned short* featBf = (unsigned short*)(ws + SZ_BANDS);
  unsigned short* tmpBf  = (unsigned short*)(ws + SZ_BANDS + SZ_FPAD);
  // phase 2: RWKV stage (after proj has consumed featBf) — 167,772,160 B total
  unsigned short* xkBf   = (unsigned short*)(ws + 0);
  unsigned short* xvBf   = (unsigned short*)(ws + 16777216);
  unsigned short* xrBf   = (unsigned short*)(ws + 33554432);
  float*          kF     = (float*)(ws + 50331648);
  float*          vF     = (float*)(ws + 83886080);
  float*          rF     = (float*)(ws + 117440512);
  unsigned short* rwkvBf = (unsigned short*)(ws + 150994944);

  size_t off = SZ_R0;
  auto take = [&](size_t bytes) -> size_t {
    size_t r = off;
    off += (bytes + 255) & ~(size_t)255;
    return r;
  };
  float*          hF      = (float*)(ws + take((size_t)Bsz * Tn * Hn * 4));   // 33.5 MB
  float*          outresF = (float*)(ws + take((size_t)Bsz * Tn * Hn * 4));   // 33.5 MB
  unsigned short* wtResBf = (unsigned short*)(ws + take((size_t)NBLKn * 2 * 3 * C5n * C5n * 2));
  float*          bnScale = (float*)(ws + take(NBLKn * 2 * C5n * 4));
  float*          bnBias  = (float*)(ws + take(NBLKn * 2 * C5n * 4));
  unsigned short* projWBf = (unsigned short*)(ws + take(Hn * C5n * 2));
  unsigned short* wkBf    = (unsigned short*)(ws + take(Lw * Hn * Hn * 2));
  unsigned short* wvBf    = (unsigned short*)(ws + take(Lw * Hn * Hn * 2));
  unsigned short* wrBf    = (unsigned short*)(ws + take(Lw * Hn * Hn * 2));
  unsigned short* woBf    = (unsigned short*)(ws + take(Lw * Hn * Hn * 2));
  unsigned short* cls1Bf  = (unsigned short*)(ws + take(256 * Hn * 2));
  unsigned short* cls2Bf  = (unsigned short*)(ws + take(NCLSn * 256 * 2));
  float*          pooledF = (float*)(ws + take(Bsz * Cn * 4));
  float*          attnF   = (float*)(ws + take(Bsz * Cn * 4));
  float*          xmF     = (float*)(ws + take(Bsz * Hn * 4));
  float*          cpF     = (float*)(ws + take(Bsz * Hn * 4));
  unsigned short* cpBf    = (unsigned short*)(ws + take(Bsz * Hn * 2));
  float*          hidF    = (float*)(ws + take(Bsz * 256 * 4));
  unsigned short* hidBf   = (unsigned short*)(ws + take(Bsz * 256 * 2));
  (void)ws_size; (void)in_sizes; (void)n_in; (void)out_size;

  // ---- weight prep ----
  prep_resw<<<(NBLKn * 2 * C5n * C5n * 3 + 255) / 256, 256, 0, stream>>>(res_w, wtResBf);
  prep_bn<<<(NBLKn * 2 * C5n + 255) / 256, 256, 0, stream>>>(bn_g, bn_b, bn_m, bn_v, bnScale, bnBias);
  cvt_bf16<<<(Hn * C5n + 255) / 256, 256, 0, stream>>>(proj_w, projWBf, Hn * C5n);
  cvt_bf16<<<(Lw * Hn * Hn + 255) / 256, 256, 0, stream>>>(wk, wkBf, Lw * Hn * Hn);
  cvt_bf16<<<(Lw * Hn * Hn + 255) / 256, 256, 0, stream>>>(wv, wvBf, Lw * Hn * Hn);
  cvt_bf16<<<(Lw * Hn * Hn + 255) / 256, 256, 0, stream>>>(wr, wrBf, Lw * Hn * Hn);
  cvt_bf16<<<(Lw * Hn * Hn + 255) / 256, 256, 0, stream>>>(wo, woBf, Lw * Hn * Hn);
  cvt_bf16<<<(256 * Hn + 255) / 256, 256, 0, stream>>>(cls_w1, cls1Bf, 256 * Hn);
  cvt_bf16<<<(NCLSn * 256 + 255) / 256, 256, 0, stream>>>(cls_w2, cls2Bf, NCLSn * 256);
  pad_zero<<<(Bsz * C5n * 16 + 255) / 256, 256, 0, stream>>>(featBf, tmpBf);

  // ---- band stage ----
  dwconv<<<dim3(Tn / 256, C5n, Bsz), 256, 0, stream>>>(x, band_w, band_b, bandsF);
  pool_bc<<<dim3(Cn, Bsz), 256, 0, stream>>>(bandsF, pooledF);
  attn_softmax<<<Bsz, Cn, 0, stream>>>(pooledF, attn_w, attn_b, attnF);
  scale_bands<<<(int)(((size_t)Bsz * C5n * Tn) / 256), 256, 0, stream>>>(bandsF, featBf, attnF);

  // ---- residual conv blocks (WMMA, 16x64 tiles) ----
  for (int i = 0; i < NBLKn; ++i) {
    const unsigned short* w0 = wtResBf + (size_t)(2 * i) * 3 * C5n * C5n;
    const unsigned short* w1 = wtResBf + (size_t)(2 * i + 1) * 3 * C5n * C5n;
    res_conv_gemm<false><<<dim3(Tn / 64, C5n / 16, Bsz), 32, 0, stream>>>(
        w0, featBf, bnScale + (2 * i) * C5n, bnBias + (2 * i) * C5n,
        nullptr, nullptr, tmpBf);
    res_conv_gemm<true><<<dim3(Tn / 64, C5n / 16, Bsz), 32, 0, stream>>>(
        w1, tmpBf, bnScale + (2 * i + 1) * C5n, bnBias + (2 * i + 1) * C5n,
        bandsF, bandsF, featBf);
  }

  // ---- projection to H (WMMA): h[b,t,o] from padded feat[b,i,t] ----
  gemm_bf16<4, 0, false, false, false, true><<<dim3(Tn / 64, Hn / 16, Bsz), 32, 0, stream>>>(
      projWBf, featBf + 8, proj_b, hF, nullptr,
      /*K*/ C5n, /*Mvalid*/ Hn, /*ldo*/ Hn,
      /*bsX*/ (long long)C5n * RSn, /*bsO*/ (long long)Tn * Hn,
      /*nstride*/ 1, /*kstride*/ RSn);

  // ---- RWKV layers ----
  for (int l = 0; l < Lw; ++l) {
    mean_T<<<Bsz, Hn, 0, stream>>>(hF, xmF, nullptr);
    mix_kvr<<<(int)(((size_t)Bsz * Tn * Hn) / 256), 256, 0, stream>>>(
        hF, xmF, tmk + l * Hn, tmv + l * Hn, tmr + l * Hn, xkBf, xvBf, xrBf);
    gemm_bf16<4, 1, false, true, false, false><<<dim3(Tn / 64, Hn / 16, Bsz), 32, 0, stream>>>(
        wkBf + (size_t)l * Hn * Hn, xkBf, nullptr, kF, nullptr,
        Hn, Hn, Hn, (long long)Tn * Hn, (long long)Tn * Hn, Hn, 1);
    gemm_bf16<4, 1, false, true, false, false><<<dim3(Tn / 64, Hn / 16, Bsz), 32, 0, stream>>>(
        wvBf + (size_t)l * Hn * Hn, xvBf, nullptr, vF, nullptr,
        Hn, Hn, Hn, (long long)Tn * Hn, (long long)Tn * Hn, Hn, 1);
    gemm_bf16<4, 2, false, true, false, false><<<dim3(Tn / 64, Hn / 16, Bsz), 32, 0, stream>>>(
        wrBf + (size_t)l * Hn * Hn, xrBf, nullptr, rF, nullptr,
        Hn, Hn, Hn, (long long)Tn * Hn, (long long)Tn * Hn, Hn, 1);
    rwkv_scan<<<(Bsz * Hn) / 256, 256, 0, stream>>>(kF, vF, rF, rwkvBf);
    gemm_bf16<4, 0, false, true, false, false><<<dim3(Tn / 64, Hn / 16, Bsz), 32, 0, stream>>>(
        woBf + (size_t)l * Hn * Hn, rwkvBf, nullptr, outresF, nullptr,
        Hn, Hn, Hn, (long long)Tn * Hn, (long long)Tn * Hn, Hn, 1);
    ln_ff<<<Bsz * Tn, 128, 0, stream>>>(hF, outresF,
        ln1g + l * Hn, ln1b + l * Hn, ln2g + l * Hn, ln2b + l * Hn);
  }

  // ---- classifier ----
  mean_T<<<Bsz, Hn, 0, stream>>>(hF, cpF, cpBf);
  gemm_bf16<2, 1, false, true, true, true><<<dim3(1, 256 / 16, 1), 32, 0, stream>>>(
      cls1Bf, cpBf, cls_b1, hidF, hidBf,
      /*K*/ Hn, /*Mvalid*/ 256, /*ldo*/ 256, 0, 0,
      /*nstride*/ Hn, /*kstride*/ 1);
  gemm_bf16<2, 0, true, true, false, true><<<dim3(1, 1, 1), 32, 0, stream>>>(
      cls2Bf, hidBf, cls_b2, (float*)d_out, nullptr,
      /*K*/ 256, /*Mvalid*/ NCLSn, /*ldo*/ NCLSn, 0, 0,
      /*nstride*/ 256, /*kstride*/ 1);
}
